// MultiScaleAttentionLayer_81475529605479
// MI455X (gfx1250) — matrix operand order repaired and checked by reading, not compiled, not run on hardware
//
#include <hip/hip_runtime.h>
#include <hip/hip_bf16.h>
#include <math.h>

// ---------------------------------------------------------------------------
// MultiScaleAttentionLayer on gfx1250 (MI455X): bf16 WMMA pipeline.
//   B=2, S=2048, D=1024, H=16, dk=64
// Stages:
//  1) convert x -> bf16                                  (Xb)
//  2) transpose+convert Wq/Wk/Wv/Wo -> bf16 [N][K]       (W*t)
//  3) WMMA GEMM (64x32/wave, spill-free): Q,K -> [B][H][S][dk] ; V -> [B][H][dk][S]
//  4) flash attention, 16 q-rows/wave, fp32 online softmax
//  5) WMMA GEMM: ctx @ Wo + bo + x  -> y fp32
//  6) LayerNorm(y) -> out fp32
// ---------------------------------------------------------------------------

typedef __attribute__((ext_vector_type(16))) __bf16 v16bf;
typedef __attribute__((ext_vector_type(8)))  __bf16 bf16x8;
typedef __attribute__((ext_vector_type(8)))  float  v8f;

#define SEQ 2048
#define DM  1024
#define NH  16
#define DK  64

// Load a 16-element bf16 fragment from two contiguous 16B runs (2x b128).
static __device__ __forceinline__ v16bf ld_frag16(const __bf16* p0, const __bf16* p1) {
  bf16x8 lo = *(const bf16x8*)p0;
  bf16x8 hi = *(const bf16x8*)p1;
  v16bf r;
#pragma unroll
  for (int i = 0; i < 8; ++i) { r[i] = lo[i]; r[i + 8] = hi[i]; }
  return r;
}

static __device__ __forceinline__ v8f wmma_bf16(v16bf a, v16bf b, v8f c) {
  // (neg_a, A, neg_b, B, c_mod, C, reuse_a, reuse_b)
  return __builtin_amdgcn_wmma_f32_16x16x32_bf16(false, a, false, b, (short)0, c, false, false);
}

// ---------------------------- stage 1: convert x ---------------------------
__global__ void cvt_f32_bf16(const float* __restrict__ in, __bf16* __restrict__ out, int n) {
  int i = blockIdx.x * blockDim.x + threadIdx.x;
  if (i < n) out[i] = (__bf16)in[i];
}

// ------------------- stage 2: transpose + convert weights ------------------
// W [K][N] fp32 row-major  ->  Wt [N][K] bf16 row-major
__global__ void transpose_cvt(const float* __restrict__ W, __bf16* __restrict__ Wt) {
  __shared__ float tile[32][33];
  int n0 = blockIdx.x * 32, k0 = blockIdx.y * 32;
  for (int j = threadIdx.y; j < 32; j += 8)
    tile[j][threadIdx.x] = W[(size_t)(k0 + j) * DM + n0 + threadIdx.x];
  __syncthreads();
  for (int j = threadIdx.y; j < 32; j += 8)
    Wt[(size_t)(n0 + j) * DM + k0 + threadIdx.x] = (__bf16)tile[threadIdx.x][j];
}

// --------------------- stage 3: QKV projection GEMMs -----------------------
// C[M=4096][N=1024] = A[M][K] * Bt[N][K]^T + bias
// 4 waves/WG in 2x2; each wave computes 64x32 = 4x2 WMMA tiles (spill-free).
// mode 0: scatter bf16 to [B][H][S][dk]  (Q, K)
// mode 1: packed  bf16 to [B][H][dk][S]  (V transposed; per-lane b128 store)
__global__ void gemm_proj(const __bf16* __restrict__ A, const __bf16* __restrict__ Bt,
                          const float* __restrict__ bias, __bf16* __restrict__ out, int mode) {
  const int K = DM;
  int lane = threadIdx.x & 31, wave = threadIdx.x >> 5;
  int l16 = lane & 15, hl = lane >> 4;
  int m_wave = blockIdx.y * 128 + (wave >> 1) * 64;
  int n_wave = blockIdx.x * 64 + (wave & 1) * 32;

  v8f acc[4][2] = {};
  for (int kk = 0; kk < K; kk += 32) {
    v16bf af[4], bfg[2];
#pragma unroll
    for (int mt = 0; mt < 4; ++mt) {
      const __bf16* ar = A + (size_t)(m_wave + mt * 16 + l16) * K + kk;
      af[mt] = ld_frag16(ar + hl * 16, ar + hl * 16 + 8);
    }
#pragma unroll
    for (int nt = 0; nt < 2; ++nt) {
      const __bf16* br = Bt + (size_t)(n_wave + nt * 16 + l16) * K + kk;
      __builtin_prefetch(br + 32, 0, 1);
      bfg[nt] = ld_frag16(br + hl * 16, br + hl * 16 + 8);
    }
#pragma unroll
    for (int mt = 0; mt < 4; ++mt)
#pragma unroll
      for (int nt = 0; nt < 2; ++nt)
        acc[mt][nt] = wmma_bf16(af[mt], bfg[nt], acc[mt][nt]);
  }

#pragma unroll
  for (int mt = 0; mt < 4; ++mt) {
#pragma unroll
    for (int nt = 0; nt < 2; ++nt) {
      int n = n_wave + nt * 16 + l16;
      float bn = bias[n];
      int hh = n >> 6, dd = n & 63;
      if (mode == 0) {
#pragma unroll
        for (int r = 0; r < 8; ++r) {
          int m = m_wave + mt * 16 + 8 * hl + r;
          int bb = m >> 11, ss = m & (SEQ - 1);
          out[(((size_t)bb * NH + hh) * SEQ + ss) * DK + dd] = (__bf16)(acc[mt][nt][r] + bn);
        }
      } else {
        bf16x8 pk;
#pragma unroll
        for (int r = 0; r < 8; ++r) pk[r] = (__bf16)(acc[mt][nt][r] + bn);
        int m0 = m_wave + mt * 16 + 8 * hl;
        int bb = m0 >> 11, ss = m0 & (SEQ - 1);
        *(bf16x8*)(out + (((size_t)bb * NH + hh) * DK + dd) * SEQ + ss) = pk;
      }
    }
  }
}

// ------------------------- stage 4: flash attention ------------------------
// grid (S/64, H, B), block 128 (4 waves, each owns 16 query rows)
__global__ void attn_fwd(const __bf16* __restrict__ Qb, const __bf16* __restrict__ Kb,
                         const __bf16* __restrict__ Vt, const float* __restrict__ scale_factors,
                         __bf16* __restrict__ ctx) {
  __shared__ __bf16 plds[4][16 * 32];  // per-wave 16x32 bf16 prob tile (C->A relayout)
  int lane = threadIdx.x & 31, wave = threadIdx.x >> 5;
  int l16 = lane & 15, hl = lane >> 4;
  int b = blockIdx.z, h = blockIdx.y;
  int q0 = blockIdx.x * 64 + wave * 16;
  const __bf16* Qh = Qb + ((size_t)b * NH + h) * SEQ * DK;
  const __bf16* Kh = Kb + ((size_t)b * NH + h) * SEQ * DK;
  const __bf16* Vh = Vt + ((size_t)b * NH + h) * DK * SEQ;

  // Q fragments (held in registers across whole key loop): 16 rows x 64 dk
  const __bf16* qrow = Qh + (size_t)(q0 + l16) * DK;
  v16bf qa0 = ld_frag16(qrow + hl * 16, qrow + hl * 16 + 8);
  v16bf qa1 = ld_frag16(qrow + 32 + hl * 16, qrow + 32 + hl * 16 + 8);

  float inv_scale = 1.0f / (scale_factors[h] * 8.0f);  // 8 = sqrt(dk)

  float mrun[8], lrun[8], fac[8];
  v8f o[4] = {};
#pragma unroll
  for (int r = 0; r < 8; ++r) { mrun[r] = -1e30f; lrun[r] = 0.0f; }

  for (int key0 = 0; key0 < SEQ; key0 += 32) {
    // scores: two 16x16 tiles (keys key0..+16, +16..+32), contraction over dk=64
    v8f s0 = {}, s1 = {};
    {
      const __bf16* kr = Kh + (size_t)(key0 + l16) * DK;
      __builtin_prefetch(kr + 32 * DK, 0, 1);  // next key block
      s0 = wmma_bf16(qa0, ld_frag16(kr + hl * 16, kr + hl * 16 + 8), s0);
      s0 = wmma_bf16(qa1, ld_frag16(kr + 32 + hl * 16, kr + 32 + hl * 16 + 8), s0);
    }
    {
      const __bf16* kr = Kh + (size_t)(key0 + 16 + l16) * DK;
      s1 = wmma_bf16(qa0, ld_frag16(kr + hl * 16, kr + hl * 16 + 8), s1);
      s1 = wmma_bf16(qa1, ld_frag16(kr + 32 + hl * 16, kr + 32 + hl * 16 + 8), s1);
    }

    // online softmax; row (M) values live across the 16-lane column groups
#pragma unroll
    for (int r = 0; r < 8; ++r) {
      float v0 = s0[r] * inv_scale;
      float v1 = s1[r] * inv_scale;
      float mx = fmaxf(v0, v1);
#pragma unroll
      for (int msk = 1; msk < 16; msk <<= 1) mx = fmaxf(mx, __shfl_xor(mx, msk, 32));
      float mnew = fmaxf(mrun[r], mx);
      float p0 = __expf(v0 - mnew);
      float p1 = __expf(v1 - mnew);
      float ps = p0 + p1;
#pragma unroll
      for (int msk = 1; msk < 16; msk <<= 1) ps += __shfl_xor(ps, msk, 32);
      fac[r] = __expf(mrun[r] - mnew);
      lrun[r] = lrun[r] * fac[r] + ps;
      mrun[r] = mnew;
      int mrow = r + 8 * hl;
      plds[wave][mrow * 32 + l16]      = (__bf16)p0;  // C-layout -> row-major LDS
      plds[wave][mrow * 32 + 16 + l16] = (__bf16)p1;
    }

    // rescale running output accumulators
#pragma unroll
    for (int dt = 0; dt < 4; ++dt)
#pragma unroll
      for (int r = 0; r < 8; ++r) o[dt][r] *= fac[r];

    // reload probs as an A fragment (16x32) — wave-local LDS, in-order
    const __bf16* prow = &plds[wave][l16 * 32];
    v16bf pa = ld_frag16(prow + hl * 16, prow + hl * 16 + 8);

    // P x V : V stored transposed [dk][S] so B fragments are contiguous
#pragma unroll
    for (int dt = 0; dt < 4; ++dt) {
      const __bf16* vr = Vh + (size_t)(dt * 16 + l16) * SEQ + key0;
      v16bf vb = ld_frag16(vr + hl * 16, vr + hl * 16 + 8);
      o[dt] = wmma_bf16(pa, vb, o[dt]);
    }
  }

  // normalize and write ctx [B][S][D] bf16
#pragma unroll
  for (int dt = 0; dt < 4; ++dt) {
    int d = h * DK + dt * 16 + l16;
#pragma unroll
    for (int r = 0; r < 8; ++r) {
      int srow = q0 + r + 8 * hl;
      ctx[((size_t)b * SEQ + srow) * DM + d] = (__bf16)(o[dt][r] / lrun[r]);
    }
  }
}

// ---------------- stage 5: output projection + residual (fp32 y) -----------
__global__ void gemm_out_res(const __bf16* __restrict__ A, const __bf16* __restrict__ Bt,
                             const float* __restrict__ bias, const float* __restrict__ xres,
                             float* __restrict__ y) {
  const int K = DM;
  int lane = threadIdx.x & 31, wave = threadIdx.x >> 5;
  int l16 = lane & 15, hl = lane >> 4;
  int m_wave = blockIdx.y * 128 + (wave >> 1) * 64;
  int n_wave = blockIdx.x * 64 + (wave & 1) * 32;

  v8f acc[4][2] = {};
  for (int kk = 0; kk < K; kk += 32) {
    v16bf af[4], bfg[2];
#pragma unroll
    for (int mt = 0; mt < 4; ++mt) {
      const __bf16* ar = A + (size_t)(m_wave + mt * 16 + l16) * K + kk;
      af[mt] = ld_frag16(ar + hl * 16, ar + hl * 16 + 8);
    }
#pragma unroll
    for (int nt = 0; nt < 2; ++nt) {
      const __bf16* br = Bt + (size_t)(n_wave + nt * 16 + l16) * K + kk;
      __builtin_prefetch(br + 32, 0, 1);
      bfg[nt] = ld_frag16(br + hl * 16, br + hl * 16 + 8);
    }
#pragma unroll
    for (int mt = 0; mt < 4; ++mt)
#pragma unroll
      for (int nt = 0; nt < 2; ++nt)
        acc[mt][nt] = wmma_bf16(af[mt], bfg[nt], acc[mt][nt]);
  }

#pragma unroll
  for (int mt = 0; mt < 4; ++mt) {
#pragma unroll
    for (int nt = 0; nt < 2; ++nt) {
      int n = n_wave + nt * 16 + l16;
      float bn = bias[n];
#pragma unroll
      for (int r = 0; r < 8; ++r) {
        int m = m_wave + mt * 16 + 8 * hl + r;
        size_t idx = (size_t)m * DM + n;
        y[idx] = acc[mt][nt][r] + bn + xres[idx];
      }
    }
  }
}

// --------------------------- stage 6: LayerNorm ----------------------------
__global__ void layernorm_k(const float* __restrict__ y, const float* __restrict__ gamma,
                            const float* __restrict__ beta, float* __restrict__ out) {
  int row = blockIdx.x;
  const float* yr = y + (size_t)row * DM;
  float s = 0.f, s2 = 0.f;
  for (int i = threadIdx.x; i < DM; i += 256) { float v = yr[i]; s += v; s2 += v * v; }
#pragma unroll
  for (int msk = 1; msk < 32; msk <<= 1) { s += __shfl_xor(s, msk, 32); s2 += __shfl_xor(s2, msk, 32); }
  __shared__ float rs[8], rs2[8];
  int wave = threadIdx.x >> 5, lane = threadIdx.x & 31;
  if (lane == 0) { rs[wave] = s; rs2[wave] = s2; }
  __syncthreads();
  s = 0.f; s2 = 0.f;
#pragma unroll
  for (int w = 0; w < 8; ++w) { s += rs[w]; s2 += rs2[w]; }
  float mu = s * (1.0f / DM);
  float var = s2 * (1.0f / DM) - mu * mu;
  float rstd = rsqrtf(var + 1e-5f);
  for (int i = threadIdx.x; i < DM; i += 256)
    out[(size_t)row * DM + i] = (yr[i] - mu) * rstd * gamma[i] + beta[i];
}

// ------------------------------- launcher ----------------------------------
extern "C" void kernel_launch(void* const* d_in, const int* in_sizes, int n_in,
                              void* d_out, int out_size, void* d_ws, size_t ws_size,
                              hipStream_t stream) {
  (void)in_sizes; (void)n_in; (void)out_size; (void)ws_size;
  const float* x     = (const float*)d_in[0];
  const float* Wq    = (const float*)d_in[1];
  const float* bq    = (const float*)d_in[2];
  const float* Wk    = (const float*)d_in[3];
  const float* bk    = (const float*)d_in[4];
  const float* Wv    = (const float*)d_in[5];
  const float* bv    = (const float*)d_in[6];
  const float* Wo    = (const float*)d_in[7];
  const float* bo    = (const float*)d_in[8];
  const float* sf    = (const float*)d_in[9];
  const float* gamma = (const float*)d_in[10];
  const float* beta  = (const float*)d_in[11];
  float* out = (float*)d_out;

  char* ws = (char*)d_ws;
  const size_t MB = 1024ull * 1024ull;
  __bf16* Xb  = (__bf16*)(ws + 0);        //  8 MB  x in bf16 [4096][1024]
  __bf16* Wqt = (__bf16*)(ws + 8  * MB);  //  2 MB  Wq^T bf16
  __bf16* Wkt = (__bf16*)(ws + 10 * MB);  //  2 MB
  __bf16* Wvt = (__bf16*)(ws + 12 * MB);  //  2 MB
  __bf16* Wot = (__bf16*)(ws + 14 * MB);  //  2 MB
  __bf16* Qb  = (__bf16*)(ws + 16 * MB);  //  8 MB  [B][H][S][dk]
  __bf16* Kb  = (__bf16*)(ws + 24 * MB);  //  8 MB  [B][H][S][dk]
  __bf16* Vt  = (__bf16*)(ws + 32 * MB);  //  8 MB  [B][H][dk][S]
  __bf16* Ctx = (__bf16*)(ws + 40 * MB);  //  8 MB  [B][S][D]
  float*  Y   = (float*)(ws + 48 * MB);   // 16 MB  pre-LN fp32

  int nx = 2 * SEQ * DM;
  cvt_f32_bf16<<<nx / 256, 256, 0, stream>>>(x, Xb, nx);
  dim3 tb(32, 8), tg(DM / 32, DM / 32);
  transpose_cvt<<<tg, tb, 0, stream>>>(Wq, Wqt);
  transpose_cvt<<<tg, tb, 0, stream>>>(Wk, Wkt);
  transpose_cvt<<<tg, tb, 0, stream>>>(Wv, Wvt);
  transpose_cvt<<<tg, tb, 0, stream>>>(Wo, Wot);

  dim3 gg(DM / 64, (2 * SEQ) / 128);
  gemm_proj<<<gg, 128, 0, stream>>>(Xb, Wqt, bq, Qb, 0);
  gemm_proj<<<gg, 128, 0, stream>>>(Xb, Wkt, bk, Kb, 0);
  gemm_proj<<<gg, 128, 0, stream>>>(Xb, Wvt, bv, Vt, 1);

  dim3 ga(SEQ / 64, NH, 2);
  attn_fwd<<<ga, 128, 0, stream>>>(Qb, Kb, Vt, sf, Ctx);

  gemm_out_res<<<gg, 128, 0, stream>>>(Ctx, Wot, bo, x, Y);

  layernorm_k<<<2 * SEQ, 256, 0, stream>>>(Y, gamma, beta, out);
}
